// RNN_stack_67061619360049
// MI455X (gfx1250) — matrix operand order repaired and checked by reading, not compiled, not run on hardware
//
#include <hip/hip_runtime.h>

#define B_     32
#define T_     512
#define VOCAB_ 1024
#define D_     256
#define H_     256
#define O_     1024

typedef __attribute__((ext_vector_type(16))) _Float16 v16h;
typedef __attribute__((ext_vector_type(8)))  _Float16 v8h;
typedef __attribute__((ext_vector_type(8)))  float    v8f;
typedef __attribute__((ext_vector_type(4)))  float    f4;

// ---------- f16 fragment loaders (ISA 16-bit A 16x32 / B 32x16 lane layouts) ----------
// A: lane = half*16 + row; elements 0..7 -> K=k0+half*8+e ; 8..15 -> K=k0+16+half*8+e
static __device__ __forceinline__ v16h load_a_h(const _Float16* __restrict__ A, int ldk,
                                                int row0, int k0, int lane) {
  int r    = row0 + (lane & 15);
  int half = lane >> 4;
  const _Float16* p = A + (size_t)r * ldk + k0 + half * 8;
  v8h lo = *(const v8h*)(p);
  v8h hi = *(const v8h*)(p + 16);
  v16h a;
#pragma unroll
  for (int e = 0; e < 8; ++e) { a[e] = lo[e]; a[8 + e] = hi[e]; }
  return a;
}
// B from weight stored [N][K] row-major: lane col = col0+(lane&15); elems 0..7 -> K=k0+half*8+e ; 8..15 -> K=k0+16+half*8+e
static __device__ __forceinline__ v16h load_b_h(const _Float16* __restrict__ Wt, int ldk,
                                                int col0, int k0, int lane) {
  int c    = col0 + (lane & 15);
  int half = lane >> 4;
  const _Float16* p = Wt + (size_t)c * ldk + k0 + half * 8;
  v8h lo = *(const v8h*)(p);
  v8h hi = *(const v8h*)(p + 16);
  v16h b;
#pragma unroll
  for (int e = 0; e < 8; ++e) { b[e] = lo[e]; b[8 + e] = hi[e]; }
  return b;
}

// ---------------- one-time f32 -> f16 pack (8 elems/thread, b128 stores) ----------------
__global__ __launch_bounds__(256) void f32_to_f16(const float* __restrict__ src,
                                                  _Float16* __restrict__ dst, int n) {
  int i = (blockIdx.x * 256 + threadIdx.x) * 8;
  if (i >= n) return;
  f4 a = *(const f4*)(src + i);
  f4 b = *(const f4*)(src + i + 4);
  v8h o;
#pragma unroll
  for (int e = 0; e < 4; ++e) { o[e] = (_Float16)a[e]; o[4 + e] = (_Float16)b[e]; }
  *(v8h*)(dst + i) = o;
}

// ============ C[M,N] = A[M,K] * W[N,K]^T, f16 in, CT out (f16 or f32), f32 acc ==========
// 256 threads = 8 waves; wave tile 16(M) x 64(N); block tile 32 x 256.
template <typename CT>
__global__ __launch_bounds__(256) void gemm_h(const _Float16* __restrict__ A,
                                              const _Float16* __restrict__ Wt,
                                              CT* __restrict__ C, int N, int K) {
  int lane = threadIdx.x & 31;
  int w    = threadIdx.x >> 5;
  int row0 = blockIdx.y * 32 + (w & 1) * 16;
  int col0 = blockIdx.x * 256 + (w >> 1) * 64;
  v8f acc[4] = {};
  for (int k0 = 0; k0 < K; k0 += 32) {
    if (k0 + 32 < K) {  // hide L2 latency for the streamed A operand
      __builtin_prefetch(A + (size_t)(row0 + (lane & 15)) * K + k0 + 32, 0, 1);
    }
    v16h a = load_a_h(A, K, row0, k0, lane);
#pragma unroll
    for (int j = 0; j < 4; ++j) {
      v16h b = load_b_h(Wt, K, col0 + j * 16, k0, lane);
      acc[j] = __builtin_amdgcn_wmma_f32_16x16x32_f16(false, a, false, b, (short)0,
                                                      acc[j], false, false);
    }
  }
  int half = lane >> 4;
#pragma unroll
  for (int j = 0; j < 4; ++j) {
    int c = col0 + j * 16 + (lane & 15);
    size_t base = (size_t)(row0 + half * 8) * N + c;
#pragma unroll
    for (int v = 0; v < 8; ++v) C[base + (size_t)v * N] = (CT)acc[j][v];
  }
}

// ================= Sequential recurrence: h_t = XW[:,t,:] + h_{t-1} @ U^T ================
// One persistent workgroup (32 waves = one WGP). U lives in VGPRs (8 B-fragments/wave),
// h lives in LDS as f16 (16 KB). 8 WMMAs + 2 barriers per step.
__global__ __launch_bounds__(1024) void recur_kernel(const _Float16* __restrict__ XW, // [B,T,H] f16
                                                     const _Float16* __restrict__ U,  // [H,H]  f16
                                                     const float* __restrict__ h0,    // [H]    f32
                                                     _Float16* __restrict__ HS) {     // [B,T,H] f16
  __shared__ __align__(32) _Float16 hh[B_ * H_];
  int lane = threadIdx.x & 31;
  int w    = threadIdx.x >> 5;
  int wm   = w & 1;    // batch-row tile: rows 0..15 / 16..31
  int wn   = w >> 1;   // H-column tile: cols wn*16 .. wn*16+15

  v16h bu[8];
#pragma unroll
  for (int kc = 0; kc < 8; ++kc) bu[kc] = load_b_h(U, H_, wn * 16, kc * 32, lane);

  for (int i = threadIdx.x; i < B_ * H_; i += 1024) hh[i] = (_Float16)h0[i & (H_ - 1)];
  __syncthreads();

  int c     = wn * 16 + (lane & 15);
  int half  = lane >> 4;
  int rbase = wm * 16 + half * 8;
  int arow  = wm * 16 + (lane & 15);

  for (int t = 0; t < T_; ++t) {
    v8f acc;
#pragma unroll
    for (int v = 0; v < 8; ++v)
      acc[v] = (float)XW[((size_t)(rbase + v) * T_ + t) * H_ + c];
#pragma unroll
    for (int kc = 0; kc < 8; ++kc) {
      const _Float16* p = hh + arow * H_ + kc * 32 + half * 8;
      v8h lo = *(const v8h*)(p);
      v8h hi = *(const v8h*)(p + 16);
      v16h a;
#pragma unroll
      for (int e = 0; e < 8; ++e) { a[e] = lo[e]; a[8 + e] = hi[e]; }
      acc = __builtin_amdgcn_wmma_f32_16x16x32_f16(false, a, false, bu[kc], (short)0,
                                                   acc, false, false);
    }
    __syncthreads();   // all waves done reading hh for this step
#pragma unroll
    for (int v = 0; v < 8; ++v) {
      int r = rbase + v;
      _Float16 hv = (_Float16)acc[v];
      HS[((size_t)r * T_ + t) * H_ + c] = hv;
      hh[r * H_ + c] = hv;
    }
    __syncthreads();   // new h visible before next step's reads
  }
}

// ================= Softmax over the T axis, in place on [B,T,O] f32 =====================
__global__ __launch_bounds__(1024) void softmax_t(float* __restrict__ out) {
  int b = blockIdx.x;
  int o = threadIdx.x;                       // O_ == 1024 threads
  float* col = out + (size_t)b * T_ * O_ + o;
  float m = -3.402823466e38f;
  for (int t = 0; t < T_; ++t) m = fmaxf(m, col[(size_t)t * O_]);
  float s = 0.f;
  for (int t = 0; t < T_; ++t) s += __expf(col[(size_t)t * O_] - m);
  float inv = 1.0f / s;
  for (int t = 0; t < T_; ++t) col[(size_t)t * O_] = __expf(col[(size_t)t * O_] - m) * inv;
}

extern "C" void kernel_launch(void* const* d_in, const int* in_sizes, int n_in,
                              void* d_out, int out_size, void* d_ws, size_t ws_size,
                              hipStream_t stream) {
  (void)n_in; (void)out_size; (void)ws_size;
  const float* x     = (const float*)d_in[0];
  const float* hid   = (const float*)d_in[1];
  const float* W_emb = (const float*)d_in[2];
  const float* Wst   = (const float*)d_in[3];
  const float* Ust   = (const float*)d_in[4];
  const float* Vst   = (const float*)d_in[5];
  const float* W_out = (const float*)d_in[6];
  const int L = in_sizes[3] / (H_ * D_);   // stack_length (device scalar; derive on host)

  const size_t MT = (size_t)B_ * T_;       // 16384 rows
  // f16 arena in d_ws
  _Float16* ar    = (_Float16*)d_ws;
  _Float16* xh    = ar;                           size_t off = MT * VOCAB_;    // x f16
  _Float16* bufA  = ar + off;                     off += MT * D_;
  _Float16* bufB  = ar + off;                     off += MT * H_;
  _Float16* bufC  = ar + off;                     off += MT * H_;
  _Float16* wembh = ar + off;                     off += (size_t)D_ * VOCAB_;
  _Float16* wh    = ar + off;                     off += (size_t)L * H_ * D_;
  _Float16* uh    = ar + off;                     off += (size_t)L * H_ * H_;
  _Float16* vh    = ar + off;                     off += (size_t)L * D_ * H_;
  _Float16* wouth = ar + off;                     off += (size_t)O_ * D_;
  float*    out   = (float*)d_out;

  dim3 blk(256);
  // one-time f32 -> f16 packs
  {
    int n = (int)(MT * VOCAB_);
    f32_to_f16<<<dim3(n / (256 * 8)), blk, 0, stream>>>(x, xh, n);
    n = D_ * VOCAB_;
    f32_to_f16<<<dim3((n + 2047) / 2048), blk, 0, stream>>>(W_emb, wembh, n);
    n = L * H_ * D_;
    f32_to_f16<<<dim3((n + 2047) / 2048), blk, 0, stream>>>(Wst, wh, n);
    f32_to_f16<<<dim3((n + 2047) / 2048), blk, 0, stream>>>(Ust, uh, n);
    f32_to_f16<<<dim3((n + 2047) / 2048), blk, 0, stream>>>(Vst, vh, n);
    n = O_ * D_;
    f32_to_f16<<<dim3((n + 2047) / 2048), blk, 0, stream>>>(W_out, wouth, n);
  }

  // Embedding: inp = x @ W_emb^T   (16384 x 256, K=1024)
  gemm_h<_Float16><<<dim3(D_ / 256, (int)(MT / 32)), blk, 0, stream>>>(
      xh, wembh, bufA, D_, VOCAB_);
  for (int l = 0; l < L; ++l) {
    const _Float16* Wl = wh + (size_t)l * H_ * D_;
    const _Float16* Ul = uh + (size_t)l * H_ * H_;
    const _Float16* Vl = vh + (size_t)l * D_ * H_;
    // XW = inp @ W_l^T  (h-independent term hoisted out of the scan)
    gemm_h<_Float16><<<dim3(H_ / 256, (int)(MT / 32)), blk, 0, stream>>>(
        bufA, Wl, bufB, H_, D_);
    // sequential scan over T on one WGP
    recur_kernel<<<1, 1024, 0, stream>>>(bufB, Ul, hid + (size_t)l * H_, bufC);
    // out_t = h_t @ V_l^T, batched over all t
    gemm_h<_Float16><<<dim3(D_ / 256, (int)(MT / 32)), blk, 0, stream>>>(
        bufC, Vl, bufA, D_, H_);
  }
  // logits = inp @ W_out^T  -> f32 for softmax
  gemm_h<float><<<dim3(O_ / 256, (int)(MT / 32)), blk, 0, stream>>>(
      bufA, wouth, out, O_, D_);
  // softmax over sequence dim (axis=1), in place
  softmax_t<<<B_, O_, 0, stream>>>(out);
}
